// TissueGNNEncoder_7172595384890
// MI455X (gfx1250) — matrix operand order repaired and checked by reading, compiled, not run on hardware
//
#include <hip/hip_runtime.h>
#include <hip/hip_bf16.h>

#define HID 192
#define HEADS 4
#define CH 48
#define NQ_C 20

typedef __attribute__((ext_vector_type(16))) _Float16 v16h;
typedef __attribute__((ext_vector_type(8)))  _Float16 h8;
typedef __attribute__((ext_vector_type(8)))  float    v8f;

static inline int ceilDiv(int a, int b) { return (a + b - 1) / b; }

// ---------- monotonic float<->int encoding for atomicMax on floats ----------
__device__ __forceinline__ int f2oi(float f) {
  int i = __float_as_int(f);
  return i >= 0 ? i : (i ^ 0x7fffffff);
}
__device__ __forceinline__ float oi2f(int i) {
  return __int_as_float(i >= 0 ? i : (i ^ 0x7fffffff));
}

// ---------- utility kernels ----------
__global__ void k_fill_f32(float* p, float v, int n) {
  int i = blockIdx.x * blockDim.x + threadIdx.x;
  if (i < n) p[i] = v;
}
__global__ void k_fill_i32(int* p, int v, int n) {
  int i = blockIdx.x * blockDim.x + threadIdx.x;
  if (i < n) p[i] = v;
}
__global__ void k_cvt_h(const float* __restrict__ x, _Float16* __restrict__ y, int n) {
  int i = blockIdx.x * blockDim.x + threadIdx.x;
  if (i < n) y[i] = (_Float16)x[i];
}

// ---------- WMMA linear: Y[rows,192] = X(f16)[rows,192] @ W(f16,[192out,192in]).T (+bias) (+=Y if acc) ----------
// one wave computes one 16x16 output tile; block = 4 waves; grid = (ceil(rows/16), 3)
// - OOB A rows handled by clamping the load row (A row m only influences D row m,
//   and OOB D rows are never stored) -> zero divergence in the hot loop, b128 loads.
// - full-tile prologue/epilogue selected by a UNIFORM scalar branch (r0+16<=rows),
//   so the common case has 8 unguarded stores off one base pointer.
__global__ __launch_bounds__(128)
void k_linear_wmma(const _Float16* __restrict__ X, const _Float16* __restrict__ W,
                   const float* __restrict__ bias, float* __restrict__ Y,
                   int rows, int acc) {
  const int lane = threadIdx.x & 31;
  const int wave = threadIdx.x >> 5;
  const int r0 = blockIdx.x * 16;
  const int c0 = (blockIdx.y * 4 + wave) * 16;
  const int lr = lane & 15;
  const int hi = lane >> 4;                  // 0 or 1 (half-wave selector)
  const int arow = r0 + lr;                  // A row owned by this lane
  const int arow_c = arow < rows ? arow : rows - 1;  // clamped (garbage OK, never stored)
  const int bcol = c0 + lr;                  // output column (B column) for this lane
  const bool full = (r0 + 16) <= rows;       // uniform across the block

  const _Float16* __restrict__ Arow = X + (size_t)arow_c * HID;
  const _Float16* __restrict__ Brow = W + (size_t)bcol * HID;
  float* __restrict__ Yp = Y + (size_t)(r0 + hi * 8) * HID + c0 + lr;  // c[r] -> Yp[r*HID]

  const float cvb = bias ? bias[c0 + lr] : 0.f;
  v8f c;
  if (acc) {
    if (full) {
#pragma unroll
      for (int r = 0; r < 8; ++r) c[r] = cvb + Yp[(size_t)r * HID];
    } else {
#pragma unroll
      for (int r = 0; r < 8; ++r) {
        const int m = r0 + r + hi * 8;
        c[r] = cvb + (m < rows ? Y[(size_t)m * HID + c0 + lr] : 0.f);
      }
    }
  } else {
#pragma unroll
    for (int r = 0; r < 8; ++r) c[r] = cvb;
  }

#pragma unroll
  for (int kt = 0; kt < 6; ++kt) {           // K = 192 = 6 * 32
    const int kb = kt * 32;
    // A fragment: lane covers K {kb+hi*8 .. +7} and {kb+16+hi*8 .. +7} -> 2 x b128
    h8 a0 = *(const h8*)(Arow + kb + hi * 8);
    h8 a1 = *(const h8*)(Arow + kb + 16 + hi * 8);
    // B fragment: lane covers K {kb+hi*8 .. +7} and {kb+16+hi*8 .. +7} -> 2 x b128
    h8 b0 = *(const h8*)(Brow + kb + hi * 8);
    h8 b1 = *(const h8*)(Brow + kb + 16 + hi * 8);
    v16h a = __builtin_shufflevector(a0, a1, 0, 1, 2, 3, 4, 5, 6, 7,
                                     8, 9, 10, 11, 12, 13, 14, 15);
    v16h b = __builtin_shufflevector(b0, b1, 0, 1, 2, 3, 4, 5, 6, 7,
                                     8, 9, 10, 11, 12, 13, 14, 15);
    c = __builtin_amdgcn_wmma_f32_16x16x32_f16(false, a, false, b, (short)0, c,
                                               false, false);
  }

  if (full) {
#pragma unroll
    for (int r = 0; r < 8; ++r) Yp[(size_t)r * HID] = c[r];
  } else {
#pragma unroll
    for (int r = 0; r < 8; ++r) {
      const int m = r0 + r + hi * 8;
      if (m < rows) Y[(size_t)m * HID + c0 + lr] = c[r];
    }
  }
}

// ---------- SAGE mean aggregation ----------
__global__ void k_seg_sum(const float* __restrict__ x, const int* __restrict__ src,
                          const int* __restrict__ dst, float* __restrict__ s,
                          float* __restrict__ cnt, int E) {
  int e = blockIdx.x;
  if (e >= E) return;
  int f = threadIdx.x;
  int si = src[e], di = dst[e];
  atomicAdd(&s[(size_t)di * HID + f], x[(size_t)si * HID + f]);
  if (f == 0) atomicAdd(&cnt[di], 1.0f);
}
__global__ void k_mean_h(const float* __restrict__ s, const float* __restrict__ cnt,
                         _Float16* __restrict__ o, int total) {
  int i = blockIdx.x * blockDim.x + threadIdx.x;
  if (i >= total) return;
  float cc = cnt[i / HID];
  cc = cc > 1.f ? cc : 1.f;
  o[i] = (_Float16)(s[i] / cc);
}

// ---------- GATv2 edge kernels ----------
__global__ void k_gat_score(const float* __restrict__ gl, const float* __restrict__ gr,
                            const int* __restrict__ src, const int* __restrict__ dst,
                            const float* __restrict__ att, float* __restrict__ score,
                            int* __restrict__ mmax, int E) {
  int idx = blockIdx.x * blockDim.x + threadIdx.x;
  if (idx >= E * HEADS) return;
  int h = idx & 3;
  int e = idx >> 2;
  int si = src[e], di = dst[e];
  const float* a = gl + (size_t)si * HID + h * CH;
  const float* b = gr + (size_t)di * HID + h * CH;
  const float* w = att + h * CH;
  float acc = 0.f;
  for (int cc = 0; cc < CH; ++cc) {
    float v = a[cc] + b[cc];
    v = v > 0.f ? v : 0.2f * v;   // leaky relu(0.2)
    acc += v * w[cc];
  }
  score[idx] = acc;
  atomicMax(&mmax[di * HEADS + h], f2oi(acc));
}
__global__ void k_gat_exp(const float* __restrict__ score, const int* __restrict__ dst,
                          const int* __restrict__ mmax, float* __restrict__ den,
                          float* __restrict__ exb, int E) {
  int idx = blockIdx.x * blockDim.x + threadIdx.x;
  if (idx >= E * HEADS) return;
  int h = idx & 3;
  int e = idx >> 2;
  int di = dst[e];
  float ex = expf(score[idx] - oi2f(mmax[di * HEADS + h]));
  exb[idx] = ex;
  atomicAdd(&den[di * HEADS + h], ex);
}
__global__ void k_gat_agg(const float* __restrict__ exb, const float* __restrict__ den,
                          const float* __restrict__ gl, const int* __restrict__ src,
                          const int* __restrict__ dst, float* __restrict__ out, int E) {
  int e = blockIdx.x;
  if (e >= E) return;
  int f = threadIdx.x;
  int h = f / CH;
  int si = src[e], di = dst[e];
  float alpha = exb[e * HEADS + h] / (den[di * HEADS + h] + 1e-16f);
  atomicAdd(&out[(size_t)di * HID + f], alpha * gl[(size_t)si * HID + f]);
}
__global__ void k_add_bias(float* __restrict__ y, const float* __restrict__ b, int total) {
  int i = blockIdx.x * blockDim.x + threadIdx.x;
  if (i < total) y[i] += b[i % HID];
}

// ---------- residual + LayerNorm (+ fp16 mirror) ----------
__global__ __launch_bounds__(64)
void k_ln(float* __restrict__ x, const float* __restrict__ delta,
          const float* __restrict__ g, const float* __restrict__ b,
          _Float16* __restrict__ xh, int nrows) {
  int row = blockIdx.x;
  if (row >= nrows) return;
  int t = threadIdx.x;
  __shared__ float red[64];
  float v[3];
  float s = 0.f;
#pragma unroll
  for (int j = 0; j < 3; ++j) {
    int f = t + j * 64;
    float xv = x[(size_t)row * HID + f];
    if (delta) xv += delta[(size_t)row * HID + f];
    v[j] = xv;
    s += xv;
  }
  red[t] = s; __syncthreads();
  for (int o = 32; o > 0; o >>= 1) { if (t < o) red[t] += red[t + o]; __syncthreads(); }
  float mu = red[0] * (1.f / HID);
  __syncthreads();
  float s2 = 0.f;
#pragma unroll
  for (int j = 0; j < 3; ++j) { float d = v[j] - mu; s2 += d * d; }
  red[t] = s2; __syncthreads();
  for (int o = 32; o > 0; o >>= 1) { if (t < o) red[t] += red[t + o]; __syncthreads(); }
  float rinv = rsqrtf(red[0] * (1.f / HID) + 1e-5f);
#pragma unroll
  for (int j = 0; j < 3; ++j) {
    int f = t + j * 64;
    float y = (v[j] - mu) * rinv * g[f] + b[f];
    x[(size_t)row * HID + f] = y;
    if (xh) xh[(size_t)row * HID + f] = (_Float16)y;
  }
}

// ---------- pooling attention ----------
__global__ void k_attn_scores(const float* __restrict__ q, const float* __restrict__ k,
                              float* __restrict__ sc, int n) {
  int idx = blockIdx.x * blockDim.x + threadIdx.x;
  if (idx >= HEADS * NQ_C * n) return;
  int nn = idx % n;
  int qi = (idx / n) % NQ_C;
  int h = idx / (n * NQ_C);
  const float* qp = q + (size_t)qi * HID + h * CH;
  const float* kp = k + (size_t)nn * HID + h * CH;
  float a = 0.f;
  for (int cc = 0; cc < CH; ++cc) a += qp[cc] * kp[cc];
  sc[idx] = a * 0.14433756729740643f;   // 1/sqrt(48)
}
__global__ __launch_bounds__(256)
void k_attn_softmax(float* __restrict__ sc, int n) {   // n == 512
  int base = blockIdx.x * n;
  int t = threadIdx.x;
  __shared__ float red[256];
  float a = sc[base + t];
  float b = sc[base + t + 256];
  red[t] = fmaxf(a, b); __syncthreads();
  for (int o = 128; o > 0; o >>= 1) { if (t < o) red[t] = fmaxf(red[t], red[t + o]); __syncthreads(); }
  float mx = red[0]; __syncthreads();
  float ea = expf(a - mx), eb = expf(b - mx);
  red[t] = ea + eb; __syncthreads();
  for (int o = 128; o > 0; o >>= 1) { if (t < o) red[t] += red[t + o]; __syncthreads(); }
  float inv = 1.f / red[0];
  sc[base + t] = ea * inv;
  sc[base + t + 256] = eb * inv;
}
__global__ void k_attn_out(const float* __restrict__ sc, const float* __restrict__ v,
                           float* __restrict__ o, int n) {
  int idx = blockIdx.x * blockDim.x + threadIdx.x;
  if (idx >= NQ_C * HID) return;
  int f = idx % HID;
  int qi = idx / HID;
  int h = f / CH;
  const float* sp = sc + (size_t)(h * NQ_C + qi) * n;
  float a = 0.f;
  for (int i = 0; i < n; ++i) a += sp[i] * v[(size_t)i * HID + f];
  o[idx] = a;
}
__global__ void k_qmean(const float* __restrict__ om, float* __restrict__ pooled) {
  int f = blockIdx.x * blockDim.x + threadIdx.x;
  if (f >= HID) return;
  float a = 0.f;
  for (int q = 0; q < NQ_C; ++q) a += om[q * HID + f];
  pooled[f] = a * (1.f / NQ_C);
}
__global__ void k_matvec(const float* __restrict__ W, const float* __restrict__ b,
                         const float* __restrict__ x, float* __restrict__ y,
                         int outn, int relu) {
  int o = blockIdx.x * blockDim.x + threadIdx.x;
  if (o >= outn) return;
  float a = b ? b[o] : 0.f;
  for (int k = 0; k < HID; ++k) a += W[(size_t)o * HID + k] * x[k];
  if (relu) a = fmaxf(a, 0.f);
  y[o] = a;
}

// ---------- host orchestration ----------
extern "C" void kernel_launch(void* const* d_in, const int* in_sizes, int n_in,
                              void* d_out, int out_size, void* d_ws, size_t ws_size,
                              hipStream_t stream) {
  (void)n_in; (void)out_size; (void)ws_size;
  const float* in_xs     = (const float*)d_in[0];
  const float* in_xg     = (const float*)d_in[1];
  const float* in_xp     = (const float*)d_in[2];
  const int* e_sg_src    = (const int*)d_in[3];
  const int* e_sg_dst    = (const int*)d_in[4];
  const int* e_gp_src    = (const int*)d_in[5];
  const int* e_gp_dst    = (const int*)d_in[6];
  const int* e_pp_src    = (const int*)d_in[7];
  const int* e_pp_dst    = (const int*)d_in[8];
  const float* sage_Wl   = (const float*)d_in[9];
  const float* sage_bl   = (const float*)d_in[10];
  const float* sage_Wr   = (const float*)d_in[11];
  const float* gat_Wl    = (const float*)d_in[12];
  const float* gat_bl    = (const float*)d_in[13];
  const float* gat_Wr    = (const float*)d_in[14];
  const float* gat_br    = (const float*)d_in[15];
  const float* gat_att   = (const float*)d_in[16];
  const float* gat_bias  = (const float*)d_in[17];
  const float* ln_g      = (const float*)d_in[18];
  const float* ln_b      = (const float*)d_in[19];
  const float* pool_query= (const float*)d_in[20];
  const float* pool_in_w = (const float*)d_in[21];
  const float* pool_in_b = (const float*)d_in[22];
  const float* pool_out_w= (const float*)d_in[23];
  const float* pool_out_b= (const float*)d_in[24];
  const float* ff_w1     = (const float*)d_in[25];
  const float* ff_b1     = (const float*)d_in[26];
  const float* ff_w2     = (const float*)d_in[27];
  const float* ff_b2     = (const float*)d_in[28];
  const float* pool_ln_g = (const float*)d_in[29];
  const float* pool_ln_b = (const float*)d_in[30];
  const float* proj_w    = (const float*)d_in[31];
  const float* proj_b    = (const float*)d_in[32];

  const int NSMP = in_sizes[0] / HID;      // 512
  const int NGEN = in_sizes[1] / HID;      // 11560
  const int NPTH = in_sizes[2] / HID;      // 2000
  const int E_SG = in_sizes[3];            // 1e6
  const int E_GP = in_sizes[5];            // 2e5
  const int E_PP = in_sizes[7];            // 5e4
  const int MAT  = HID * HID;              // 36864

  // ---- bump allocator over d_ws ----
  char* wp = (char*)d_ws;
  auto afl = [&](size_t n) -> float* {
    float* r = (float*)wp; wp += ((n * 4 + 255) / 256) * 256; return r;
  };
  auto ahf = [&](size_t n) -> _Float16* {
    _Float16* r = (_Float16*)wp; wp += ((n * 2 + 255) / 256) * 256; return r;
  };

  float* xs    = afl((size_t)NSMP * HID);
  float* xg    = afl((size_t)NGEN * HID);
  float* xp    = afl((size_t)NPTH * HID);
  float* og    = afl((size_t)NGEN * HID);
  float* os_   = afl((size_t)NSMP * HID);
  float* op    = afl((size_t)NPTH * HID);
  float* aggs  = afl((size_t)NGEN * HID);
  float* cnt   = afl((size_t)NGEN);
  float* gl    = afl((size_t)NGEN * HID);
  float* gr    = afl((size_t)NGEN * HID);
  float* score = afl((size_t)E_GP * HEADS);
  float* exb   = afl((size_t)E_GP * HEADS);
  int*   mmax  = (int*)afl((size_t)NGEN * HEADS);
  float* den   = afl((size_t)NGEN * HEADS);
  float* kmat  = afl((size_t)NSMP * HID);
  float* vmat  = afl((size_t)NSMP * HID);
  float* qmat  = afl((size_t)NQ_C * HID);
  float* sc    = afl((size_t)HEADS * NQ_C * NSMP);
  float* om    = afl((size_t)NQ_C * HID);
  float* pooled= afl(HID);
  float* h1    = afl(HID);
  float* h2    = afl(HID);

  _Float16* xs_h  = ahf((size_t)NSMP * HID);
  _Float16* xg_h  = ahf((size_t)NGEN * HID);
  _Float16* xp_h  = ahf((size_t)NPTH * HID);
  _Float16* meanh = ahf((size_t)NGEN * HID);
  _Float16* sWl_h = ahf((size_t)4 * MAT);
  _Float16* sWr_h = ahf((size_t)4 * MAT);
  _Float16* gWl_h = ahf((size_t)6 * MAT);
  _Float16* gWr_h = ahf((size_t)6 * MAT);
  _Float16* pin_h = ahf((size_t)3 * MAT);
  _Float16* pout_h= ahf((size_t)MAT);
  _Float16* pq_h  = ahf((size_t)NQ_C * HID);
  _Float16* om_h  = ahf((size_t)NQ_C * HID);

  auto fillf = [&](float* p, float v, int n) {
    k_fill_f32<<<ceilDiv(n, 256), 256, 0, stream>>>(p, v, n);
  };
  auto filli = [&](int* p, int v, int n) {
    k_fill_i32<<<ceilDiv(n, 256), 256, 0, stream>>>(p, v, n);
  };
  auto cvt = [&](const float* s, _Float16* d, int n) {
    k_cvt_h<<<ceilDiv(n, 256), 256, 0, stream>>>(s, d, n);
  };
  auto lin = [&](const _Float16* X, const _Float16* W, const float* bias, float* Y,
                 int rows, int acc) {
    dim3 g(ceilDiv(rows, 16), 3);
    k_linear_wmma<<<g, 128, 0, stream>>>(X, W, bias, Y, rows, acc);
  };
  auto sage = [&](const float* xsrc, const _Float16* xdst_h, int n_dst,
                  const int* src, const int* dst, int E,
                  const _Float16* Wl, const float* bl, const _Float16* Wr, float* y) {
    fillf(aggs, 0.f, n_dst * HID);
    fillf(cnt, 0.f, n_dst);
    k_seg_sum<<<E, HID, 0, stream>>>(xsrc, src, dst, aggs, cnt, E);
    k_mean_h<<<ceilDiv(n_dst * HID, 256), 256, 0, stream>>>(aggs, cnt, meanh, n_dst * HID);
    lin(meanh, Wl, bl, y, n_dst, 0);       // y  = mean @ Wl.T + bl
    lin(xdst_h, Wr, nullptr, y, n_dst, 1); // y += x_dst @ Wr.T
  };
  auto gat = [&](const _Float16* xsrc_h, const _Float16* xdst_h, int n_src, int n_dst,
                 const int* src, const int* dst, int E,
                 const _Float16* Wl, const float* bl, const _Float16* Wr, const float* br,
                 const float* att, const float* bias, float* out) {
    lin(xsrc_h, Wl, bl, gl, n_src, 0);
    lin(xdst_h, Wr, br, gr, n_dst, 0);
    filli(mmax, (int)0x80000000, n_dst * HEADS);
    fillf(den, 0.f, n_dst * HEADS);
    k_gat_score<<<ceilDiv(E * HEADS, 256), 256, 0, stream>>>(gl, gr, src, dst, att, score, mmax, E);
    k_gat_exp<<<ceilDiv(E * HEADS, 256), 256, 0, stream>>>(score, dst, mmax, den, exb, E);
    k_gat_agg<<<E, HID, 0, stream>>>(exb, den, gl, src, dst, out, E);
    k_add_bias<<<ceilDiv(n_dst * HID, 256), 256, 0, stream>>>(out, bias, n_dst * HID);
  };

  // ---- init: working copies + f16 mirrors + f16 weights ----
  hipMemcpyAsync(xs, in_xs, (size_t)NSMP * HID * 4, hipMemcpyDeviceToDevice, stream);
  hipMemcpyAsync(xg, in_xg, (size_t)NGEN * HID * 4, hipMemcpyDeviceToDevice, stream);
  hipMemcpyAsync(xp, in_xp, (size_t)NPTH * HID * 4, hipMemcpyDeviceToDevice, stream);
  cvt(in_xs, xs_h, NSMP * HID);
  cvt(in_xg, xg_h, NGEN * HID);
  cvt(in_xp, xp_h, NPTH * HID);
  cvt(sage_Wl, sWl_h, 4 * MAT);
  cvt(sage_Wr, sWr_h, 4 * MAT);
  cvt(gat_Wl, gWl_h, 6 * MAT);
  cvt(gat_Wr, gWr_h, 6 * MAT);
  cvt(pool_in_w, pin_h, 3 * MAT);
  cvt(pool_out_w, pout_h, MAT);
  cvt(pool_query, pq_h, NQ_C * HID);

  // ---- GNN layers ----
  for (int l = 0; l < 2; ++l) {
    // og = SAGE(sample -> gene)
    sage(xs, xg_h, NGEN, e_sg_src, e_sg_dst, E_SG,
         sWl_h + (size_t)(l * 2 + 0) * MAT, sage_bl + (l * 2 + 0) * HID,
         sWr_h + (size_t)(l * 2 + 0) * MAT, og);
    // os = SAGE(gene -> sample)
    sage(xg, xs_h, NSMP, e_sg_dst, e_sg_src, E_SG,
         sWl_h + (size_t)(l * 2 + 1) * MAT, sage_bl + (l * 2 + 1) * HID,
         sWr_h + (size_t)(l * 2 + 1) * MAT, os_);
    // op = GAT(gene -> pathway) + GAT(pathway -> pathway)
    fillf(op, 0.f, NPTH * HID);
    gat(xg_h, xp_h, NGEN, NPTH, e_gp_src, e_gp_dst, E_GP,
        gWl_h + (size_t)(l * 3 + 0) * MAT, gat_bl + (l * 3 + 0) * HID,
        gWr_h + (size_t)(l * 3 + 0) * MAT, gat_br + (l * 3 + 0) * HID,
        gat_att + (l * 3 + 0) * HEADS * CH, gat_bias + (l * 3 + 0) * HID, op);
    // og += GAT(pathway -> gene)
    gat(xp_h, xg_h, NPTH, NGEN, e_gp_dst, e_gp_src, E_GP,
        gWl_h + (size_t)(l * 3 + 1) * MAT, gat_bl + (l * 3 + 1) * HID,
        gWr_h + (size_t)(l * 3 + 1) * MAT, gat_br + (l * 3 + 1) * HID,
        gat_att + (l * 3 + 1) * HEADS * CH, gat_bias + (l * 3 + 1) * HID, og);
    gat(xp_h, xp_h, NPTH, NPTH, e_pp_src, e_pp_dst, E_PP,
        gWl_h + (size_t)(l * 3 + 2) * MAT, gat_bl + (l * 3 + 2) * HID,
        gWr_h + (size_t)(l * 3 + 2) * MAT, gat_br + (l * 3 + 2) * HID,
        gat_att + (l * 3 + 2) * HEADS * CH, gat_bias + (l * 3 + 2) * HID, op);
    // residual + LN (+ fp16 mirrors)
    k_ln<<<NSMP, 64, 0, stream>>>(xs, os_, ln_g + (l * 3 + 0) * HID, ln_b + (l * 3 + 0) * HID, xs_h, NSMP);
    k_ln<<<NGEN, 64, 0, stream>>>(xg, og,  ln_g + (l * 3 + 1) * HID, ln_b + (l * 3 + 1) * HID, xg_h, NGEN);
    k_ln<<<NPTH, 64, 0, stream>>>(xp, op,  ln_g + (l * 3 + 2) * HID, ln_b + (l * 3 + 2) * HID, xp_h, NPTH);
  }

  // ---- multi-query pooling ----
  lin(pq_h, pin_h + (size_t)0 * MAT, pool_in_b + 0,        qmat, NQ_C, 0);  // Q
  lin(xs_h, pin_h + (size_t)1 * MAT, pool_in_b + HID,      kmat, NSMP, 0);  // K
  lin(xs_h, pin_h + (size_t)2 * MAT, pool_in_b + 2 * HID,  vmat, NSMP, 0);  // V
  k_attn_scores<<<ceilDiv(HEADS * NQ_C * NSMP, 256), 256, 0, stream>>>(qmat, kmat, sc, NSMP);
  k_attn_softmax<<<HEADS * NQ_C, 256, 0, stream>>>(sc, NSMP);
  k_attn_out<<<ceilDiv(NQ_C * HID, 256), 256, 0, stream>>>(sc, vmat, om, NSMP);
  cvt(om, om_h, NQ_C * HID);
  lin(om_h, pout_h, pool_out_b, qmat, NQ_C, 0);   // reuse qmat for projected output
  k_qmean<<<1, 256, 0, stream>>>(qmat, pooled);

  // ---- FFN + LN + projection ----
  k_matvec<<<1, 256, 0, stream>>>(ff_w1, ff_b1, pooled, h1, HID, 1);
  k_matvec<<<1, 256, 0, stream>>>(ff_w2, ff_b2, h1, h2, HID, 0);
  k_ln<<<1, 64, 0, stream>>>(h2, nullptr, pool_ln_g, pool_ln_b, nullptr, 1);
  k_matvec<<<1, 256, 0, stream>>>(proj_w, proj_b, h2, (float*)d_out, 128, 0);
}
